// MoEHead_4217657884981
// MI455X (gfx1250) — matrix operand order repaired and checked by reading, not compiled, not run on hardware
//
#include <hip/hip_runtime.h>
#include <hip/hip_bf16.h>
#include <math.h>

#define BATCH   16384
#define D_MODEL 1024
#define HIDDEN  4096
#define NCLS    1000
#define NEXP    8
#define TOPK    2

#define TM 128          // M tile per workgroup
#define TN 128          // N tile per workgroup
#define KT 32           // K step (bf16 WMMA depth)
#define MTILES 264      // (BATCH*TOPK)/TM + NEXP  -> covers worst-case padding
#define NPAD   (MTILES * TM)   // 33792

typedef __attribute__((ext_vector_type(16))) __bf16 v16bf;
typedef __attribute__((ext_vector_type(8)))  __bf16 v8bf;
typedef __attribute__((ext_vector_type(2)))  __bf16 v2bf;
typedef __attribute__((ext_vector_type(8)))  float  v8f;
typedef __attribute__((ext_vector_type(2)))  float  v2f;

union V16U { v16bf v; v8bf h[2]; };

// packed f32x2 -> bf16x2 (RNE); lowers directly to v_cvt_pk_bf16_f32
__device__ __forceinline__ unsigned pk2bf(float a, float b) {
  v2f f; f[0] = a; f[1] = b;
  v2bf r = __builtin_convertvector(f, v2bf);
  return __builtin_bit_cast(unsigned, r);
}

__device__ __forceinline__ float gelu_exact(float v) {
  return 0.5f * v * (1.0f + erff(v * 0.7071067811865475f));
}

// ---------------------------------------------------------------- init
__global__ void moe_init(float* __restrict__ y, int* __restrict__ perm,
                         int* __restrict__ cnt, float* __restrict__ probsum) {
  int i = blockIdx.x * blockDim.x + threadIdx.x;
  if (i < BATCH * NCLS) y[i] = 0.0f;
  if (i < NPAD) perm[i] = -1;
  if (i < NEXP) { cnt[i] = 0; probsum[i] = 0.0f; }
}

// ---------------------------------------------------------------- gate
__global__ __launch_bounds__(256) void moe_gate(
    const float* __restrict__ x, const float* __restrict__ gw,
    const float* __restrict__ gb, int* __restrict__ tidx,
    float* __restrict__ tw, int* __restrict__ cnt, float* __restrict__ probsum) {
  const int wave = threadIdx.x >> 5, lane = threadIdx.x & 31;
  const int t = blockIdx.x * 8 + wave;
  const float* xr = x + (size_t)t * D_MODEL;
  float acc[NEXP];
#pragma unroll
  for (int e = 0; e < NEXP; ++e) acc[e] = 0.0f;
  for (int d = lane; d < D_MODEL; d += 32) {
    float xv = xr[d];
    const float* g = gw + (size_t)d * NEXP;
#pragma unroll
    for (int e = 0; e < NEXP; ++e) acc[e] += xv * g[e];
  }
#pragma unroll
  for (int e = 0; e < NEXP; ++e)
#pragma unroll
    for (int off = 16; off > 0; off >>= 1)
      acc[e] += __shfl_xor(acc[e], off, 32);
  if (lane == 0) {
    float p[NEXP];
    float mx = -1e30f;
#pragma unroll
    for (int e = 0; e < NEXP; ++e) { p[e] = acc[e] + gb[e]; mx = fmaxf(mx, p[e]); }
    float s = 0.0f;
#pragma unroll
    for (int e = 0; e < NEXP; ++e) { p[e] = expf(p[e] - mx); s += p[e]; }
    float inv = 1.0f / s;
#pragma unroll
    for (int e = 0; e < NEXP; ++e) p[e] *= inv;
    int i0 = 0;
#pragma unroll
    for (int e = 1; e < NEXP; ++e) if (p[e] > p[i0]) i0 = e;
    int i1 = (i0 == 0) ? 1 : 0;
#pragma unroll
    for (int e = 0; e < NEXP; ++e) if (e != i0 && p[e] > p[i1]) i1 = e;
    float v0 = p[i0], v1 = p[i1], rn = 1.0f / (v0 + v1);
    tidx[t * 2 + 0] = i0; tidx[t * 2 + 1] = i1;
    tw[t * 2 + 0] = v0 * rn; tw[t * 2 + 1] = v1 * rn;
    atomicAdd(&cnt[i0], 1); atomicAdd(&cnt[i1], 1);
#pragma unroll
    for (int e = 0; e < NEXP; ++e) atomicAdd(&probsum[e], p[e]);
  }
}

// ---------------------------------------------------------------- offsets + aux
__global__ void moe_offsets(const int* __restrict__ cnt,
                            const float* __restrict__ probsum,
                            int* __restrict__ po, int* __restrict__ fill,
                            float* __restrict__ auxout) {
  if (threadIdx.x == 0 && blockIdx.x == 0) {
    int o = 0;
    float aux = 0.0f;
    for (int e = 0; e < NEXP; ++e) {
      po[e] = o; fill[e] = o;
      o += ((cnt[e] + TM - 1) / TM) * TM;      // 128-aligned segments
      float frac = (float)cnt[e] / (float)(BATCH * TOPK);
      float imp  = probsum[e] / (float)BATCH;
      aux += frac * imp;
    }
    po[NEXP] = o;
    *auxout = (float)NEXP * aux;
  }
}

// ---------------------------------------------------------------- scatter
__global__ void moe_scatter(const int* __restrict__ tidx, const float* __restrict__ tw,
                            int* __restrict__ fill, int* __restrict__ perm,
                            float* __restrict__ permw) {
  int i = blockIdx.x * blockDim.x + threadIdx.x;   // [0, BATCH*TOPK)
  int t = i >> 1;
  int e = tidx[i];
  int pos = atomicAdd(&fill[e], 1);
  perm[pos] = t;
  permw[pos] = tw[i];
}

// ---------------------------------------------------------------- GEMM1: h = gelu(Xg @ w1[e] + b1[e])
__global__ __launch_bounds__(256) void moe_gemm1(
    const float* __restrict__ x, const float* __restrict__ w1,
    const float* __restrict__ b1, const int* __restrict__ perm,
    const int* __restrict__ po, __bf16* __restrict__ hb) {
  __shared__ __attribute__((aligned(16))) __bf16 As[TM * KT];
  __shared__ __attribute__((aligned(16))) __bf16 Bs[TN * KT];
  const int row0 = blockIdx.y * TM;
  if (row0 >= po[NEXP]) return;
  int e = 0;
#pragma unroll
  for (int i = 0; i < NEXP; ++i) if (row0 >= po[i]) e = i;
  const int n0 = blockIdx.x * TN;
  const float* W = w1 + (size_t)e * D_MODEL * HIDDEN;

  const int tid = threadIdx.x, lane = tid & 31, wave = tid >> 5;
  const int wm = wave >> 1, wn = wave & 1;        // 4x2 waves; wave tile 32Mx64N
  const int arow = tid >> 1, ahalf = tid & 1;     // A staging: 128 rows x 2 halves
  const int tokA = perm[row0 + arow];
  const float* xrow = (tokA >= 0) ? (x + (size_t)tokA * D_MODEL) : nullptr;
  const int bn = tid & (TN - 1);                  // B staging: column
  const int bkp0 = (tid >> 7) << 3;               // k-pair base: 0 or 8
  const int m = lane & 15, hig = lane >> 4;

  v8f acc[2][4] = {};

  for (int k0 = 0; k0 < D_MODEL; k0 += KT) {
    // --- stage A (gathered x rows -> packed bf16), 16 elems / thread
    uint4* dA = (uint4*)&As[arow * KT + ahalf * 16];
    if (xrow) {
      const float4* s = (const float4*)(xrow + k0 + ahalf * 16);
      float4 a0 = s[0], a1 = s[1], a2 = s[2], a3 = s[3];
      dA[0] = make_uint4(pk2bf(a0.x, a0.y), pk2bf(a0.z, a0.w),
                         pk2bf(a1.x, a1.y), pk2bf(a1.z, a1.w));
      dA[1] = make_uint4(pk2bf(a2.x, a2.y), pk2bf(a2.z, a2.w),
                         pk2bf(a3.x, a3.y), pk2bf(a3.z, a3.w));
    } else {
      dA[0] = make_uint4(0, 0, 0, 0);
      dA[1] = make_uint4(0, 0, 0, 0);
    }
    // --- stage B transposed + packed: Bs[n][k] = w1[k0+k][n0+n]
    const float* wcol = W + (size_t)k0 * HIDDEN + (n0 + bn);
#pragma unroll
    for (int kp = 0; kp < 8; ++kp) {
      const int k = (bkp0 + kp) * 2;
      float va = wcol[(size_t)k * HIDDEN];
      float vb = wcol[(size_t)(k + 1) * HIDDEN];
      *(unsigned*)&Bs[bn * KT + k] = pk2bf(va, vb);
    }
    __syncthreads();

#pragma unroll
    for (int mi = 0; mi < 2; ++mi) {
      const int mrow = wm * 32 + mi * 16 + m;
      V16U a;
      a.h[0] = *(const v8bf*)&As[mrow * KT + hig * 8];
      a.h[1] = *(const v8bf*)&As[mrow * KT + 16 + hig * 8];
#pragma unroll
      for (int ni = 0; ni < 4; ++ni) {
        const int ncol = wn * 64 + ni * 16 + m;
        V16U b;
        b.h[0] = *(const v8bf*)&Bs[ncol * KT + hig * 8];
        b.h[1] = *(const v8bf*)&Bs[ncol * KT + 16 + hig * 8];
        acc[mi][ni] = __builtin_amdgcn_wmma_f32_16x16x32_bf16(
            false, a.v, false, b.v, (short)0, acc[mi][ni], false, false);
      }
    }
    __syncthreads();
  }

  const float* b1e = b1 + (size_t)e * HIDDEN;
#pragma unroll
  for (int mi = 0; mi < 2; ++mi) {
    const int mbase = row0 + wm * 32 + mi * 16 + hig * 8;
#pragma unroll
    for (int ni = 0; ni < 4; ++ni) {
      const int col = n0 + wn * 64 + ni * 16 + m;
      const float bias = b1e[col];
#pragma unroll
      for (int j = 0; j < 8; ++j) {
        float v = acc[mi][ni][j] + bias;
        float g = gelu_exact(v);
        v2f f; f[0] = g; f[1] = g;
        v2bf r = __builtin_convertvector(f, v2bf);
        *(__bf16*)&hb[(size_t)(mbase + j) * HIDDEN + col] = r[0];
      }
    }
  }
}

// ---------------------------------------------------------------- GEMM2: y[tok] += w * (h @ w2[e] + b2[e])
__global__ __launch_bounds__(256) void moe_gemm2(
    const __bf16* __restrict__ hb, const float* __restrict__ w2,
    const float* __restrict__ b2, const int* __restrict__ perm,
    const float* __restrict__ permw, const int* __restrict__ po,
    float* __restrict__ y) {
  __shared__ __attribute__((aligned(16))) __bf16 As[TM * KT];
  __shared__ __attribute__((aligned(16))) __bf16 Bs[TN * KT];
  const int row0 = blockIdx.y * TM;
  if (row0 >= po[NEXP]) return;
  int e = 0;
#pragma unroll
  for (int i = 0; i < NEXP; ++i) if (row0 >= po[i]) e = i;
  const int n0 = blockIdx.x * TN;
  const float* W = w2 + (size_t)e * HIDDEN * NCLS;

  const int tid = threadIdx.x, lane = tid & 31, wave = tid >> 5;
  const int wm = wave >> 1, wn = wave & 1;
  const int arow = tid >> 1, ahalf = tid & 1;
  const __bf16* hrow = hb + (size_t)(row0 + arow) * HIDDEN + ahalf * 16;
  const int bn = tid & (TN - 1);
  const int bkp0 = (tid >> 7) << 3;
  const int m = lane & 15, hig = lane >> 4;

  // LDS byte offset for this thread's A-tile destination (32B chunk)
  const unsigned ldsA = (unsigned)(unsigned long long)&As[arow * KT + ahalf * 16];

  v8f acc[2][4] = {};

  for (int k0 = 0; k0 < HIDDEN; k0 += KT) {
    // --- stage A: async copy hb rows (bf16) directly into LDS, 32B / thread
    {
      const __bf16* gsrc = hrow + k0;
      asm volatile(
          "global_load_async_to_lds_b128 %0, %1, off\n\t"
          "global_load_async_to_lds_b128 %0, %1, off offset:16"
          :: "v"(ldsA), "v"(gsrc) : "memory");
    }
    // --- stage B transposed + packed, with C-edge guard
    const bool colok = (n0 + bn) < NCLS;
    const float* wcol = W + (size_t)k0 * NCLS + (n0 + bn);
#pragma unroll
    for (int kp = 0; kp < 8; ++kp) {
      const int k = (bkp0 + kp) * 2;
      float va = colok ? wcol[(size_t)k * NCLS] : 0.0f;
      float vb = colok ? wcol[(size_t)(k + 1) * NCLS] : 0.0f;
      *(unsigned*)&Bs[bn * KT + k] = pk2bf(va, vb);
    }
    asm volatile("s_wait_asynccnt 0" ::: "memory");
    __syncthreads();

#pragma unroll
    for (int mi = 0; mi < 2; ++mi) {
      const int mrow = wm * 32 + mi * 16 + m;
      V16U a;
      a.h[0] = *(const v8bf*)&As[mrow * KT + hig * 8];
      a.h[1] = *(const v8bf*)&As[mrow * KT + 16 + hig * 8];
#pragma unroll
      for (int ni = 0; ni < 4; ++ni) {
        const int ncol = wn * 64 + ni * 16 + m;
        V16U b;
        b.h[0] = *(const v8bf*)&Bs[ncol * KT + hig * 8];
        b.h[1] = *(const v8bf*)&Bs[ncol * KT + 16 + hig * 8];
        acc[mi][ni] = __builtin_amdgcn_wmma_f32_16x16x32_bf16(
            false, a.v, false, b.v, (short)0, acc[mi][ni], false, false);
      }
    }
    __syncthreads();
  }

  const float* b2e = b2 + (size_t)e * NCLS;
#pragma unroll
  for (int mi = 0; mi < 2; ++mi) {
    const int rbase = row0 + wm * 32 + mi * 16 + hig * 8;
#pragma unroll
    for (int ni = 0; ni < 4; ++ni) {
      const int col = n0 + wn * 64 + ni * 16 + m;
      if (col >= NCLS) continue;
      const float bias = b2e[col];
#pragma unroll
      for (int j = 0; j < 8; ++j) {
        const int r = rbase + j;
        const int tok = perm[r];
        if (tok >= 0) {
          atomicAdd(&y[(size_t)tok * NCLS + col],
                    permw[r] * (acc[mi][ni][j] + bias));
        }
      }
    }
  }
}

// ---------------------------------------------------------------- launcher
extern "C" void kernel_launch(void* const* d_in, const int* in_sizes, int n_in,
                              void* d_out, int out_size, void* d_ws, size_t ws_size,
                              hipStream_t stream) {
  const float* x  = (const float*)d_in[0];
  const float* gw = (const float*)d_in[1];
  const float* gb = (const float*)d_in[2];
  const float* w1 = (const float*)d_in[3];
  const float* b1 = (const float*)d_in[4];
  const float* w2 = (const float*)d_in[5];
  const float* b2 = (const float*)d_in[6];
  float* y = (float*)d_out;                      // [B*C] then aux at y[B*C]

  char* ws = (char*)d_ws;
  int*    cnt     = (int*)(ws + 0);              // 8 ints
  float*  probsum = (float*)(ws + 32);           // 8 floats
  int*    po      = (int*)(ws + 64);             // 9 ints (padded offsets)
  int*    fill    = (int*)(ws + 128);            // 8 ints
  int*    tidx    = (int*)(ws + 256);            // B*2 ints
  float*  tw      = (float*)(ws + 256 + 131072); // B*2 floats
  int*    perm    = (int*)(ws + 262400);         // NPAD ints
  float*  permw   = (float*)(ws + 397568);       // NPAD floats
  __bf16* hb      = (__bf16*)(ws + 532736);      // NPAD*HIDDEN bf16 (~277 MB)

  moe_init<<<(BATCH * NCLS + 255) / 256, 256, 0, stream>>>(y, perm, cnt, probsum);
  moe_gate<<<BATCH / 8, 256, 0, stream>>>(x, gw, gb, tidx, tw, cnt, probsum);
  moe_offsets<<<1, 32, 0, stream>>>(cnt, probsum, po, fill, y + (size_t)BATCH * NCLS);
  moe_scatter<<<(BATCH * TOPK) / 256, 256, 0, stream>>>(tidx, tw, fill, perm, permw);

  dim3 g1(HIDDEN / TN, MTILES);
  moe_gemm1<<<g1, 256, 0, stream>>>(x, w1, b1, perm, po, hb);

  dim3 g2((NCLS + TN - 1) / TN, MTILES);
  moe_gemm2<<<g2, 256, 0, stream>>>(hb, w2, b2, perm, permw, po, y);
}
